// AVRNN_85813446574513
// MI455X (gfx1250) — matrix operand order repaired and checked by reading, not compiled, not run on hardware
//
#include <hip/hip_runtime.h>
#include <math.h>

// ---------------------------------------------------------------------------
// VRNN + dense-GCN forward for MI455X (gfx1250, wave32).
// All M=1024 GEMMs run through v_wmma_f32_16x16x32_bf16 (bf16 in, f32 acc).
// K-loop is software-pipelined with register double-buffering so next-chunk
// global_load_b128s overlap the current chunk's WMMAs (partial loadcnt waits).
// ---------------------------------------------------------------------------

#define B_   1024
#define HD_  256
#define RD_  256
#define ZD_  64

typedef __attribute__((ext_vector_type(16))) __bf16 bf16x16;
typedef __attribute__((ext_vector_type(8)))  float  floatx8;

union FragAB {
    bf16x16 v;
    uint4   q[2];
};

__device__ __forceinline__ __bf16 f2bf(float f) {
    union { float f; unsigned int u; } x; x.f = f;
    unsigned int u = x.u;
    unsigned int r = u + 0x7fffu + ((u >> 16) & 1u);       // round-to-nearest-even
    if ((u & 0x7f800000u) == 0x7f800000u) r = u;           // inf/nan passthrough
    union { unsigned short s; __bf16 b; } y; y.s = (unsigned short)(r >> 16);
    return y.b;
}

// ---------------------------------------------------------------------------
// Generic WMMA bf16 GEMM:  C(f32)[M x N] (+)= A(bf16)[M x K] * Bt(bf16)[N x K]^T
//   Bt is N-major (pre-transposed weights), row stride ldbt, column offset koff.
//   Block = 256 thr = 8 waves; wave w -> rows [blk*128 + w*16, +16), cols 64.
//   ACT: 0 none, 1 leaky-relu(0.01), 2 relu.  ACC: accumulate from existing C.
//   SBF: also store bf16 row-major.  SBFT: store bf16 transposed (ldcb = M).
// ---------------------------------------------------------------------------
template<int ACT, int ACC, int BIAS, int SBF, int SBFT>
__global__ __launch_bounds__(256) void gemm_bf16_kernel(
    const __bf16* __restrict__ A, int lda,
    const __bf16* __restrict__ Bt, int ldbt, int koff,
    const float* __restrict__ bias,
    float* __restrict__ C, int ldc,
    __bf16* __restrict__ Cb, int ldcb,
    int K)
{
    const int lane   = threadIdx.x & 31;
    const int wave   = threadIdx.x >> 5;
    const int tileM  = blockIdx.x * 128 + wave * 16;
    const int tileN  = blockIdx.y * 64;
    const int mA     = lane & 15;
    const int hiHalf = lane >> 4;
    const int aSel   = hiHalf ? 16 : 0;    // A: K sub-offset per lane half
    const int nB     = lane & 15;
    const int bSel   = hiHalf ? 16 : 0;    // B: K sub-offset per lane half
    const int mBase  = hiHalf * 8;         // C/D: row base per lane half

    floatx8 acc[4];
#pragma unroll
    for (int t = 0; t < 4; ++t) {
        if (ACC) {
            const float* cp = C + (size_t)(tileM + mBase) * ldc + tileN + t * 16 + nB;
#pragma unroll
            for (int r = 0; r < 8; ++r) acc[t][r] = cp[(size_t)r * ldc];
        } else {
#pragma unroll
            for (int r = 0; r < 8; ++r) acc[t][r] = 0.0f;
        }
    }

    const __bf16* aRow  = A + (size_t)(tileM + mA) * lda + aSel;
    const __bf16* bBase = Bt + (size_t)koff + bSel;

    // ---- prologue: load chunk 0 ----
    FragAB aC, bC[4];
    aC.q[0] = *reinterpret_cast<const uint4*>(aRow);
    aC.q[1] = *reinterpret_cast<const uint4*>(aRow + 8);
#pragma unroll
    for (int t = 0; t < 4; ++t) {
        const __bf16* bp = bBase + (size_t)(tileN + t * 16 + nB) * ldbt;
        bC[t].q[0] = *reinterpret_cast<const uint4*>(bp);
        bC[t].q[1] = *reinterpret_cast<const uint4*>(bp + 8);
    }

    // ---- pipelined main loop: fetch chunk kc while multiplying chunk kc-32 --
    for (int kc = 32; kc < K; kc += 32) {
        FragAB aN, bN[4];
        const __bf16* ap = aRow + kc;
        aN.q[0] = *reinterpret_cast<const uint4*>(ap);
        aN.q[1] = *reinterpret_cast<const uint4*>(ap + 8);
#pragma unroll
        for (int t = 0; t < 4; ++t) {
            const __bf16* bp = bBase + (size_t)(tileN + t * 16 + nB) * ldbt + kc;
            bN[t].q[0] = *reinterpret_cast<const uint4*>(bp);
            bN[t].q[1] = *reinterpret_cast<const uint4*>(bp + 8);
        }
#pragma unroll
        for (int t = 0; t < 4; ++t)
            acc[t] = __builtin_amdgcn_wmma_f32_16x16x32_bf16(
                false, aC.v, false, bC[t].v, (short)0, acc[t], false, false);
        aC = aN;
#pragma unroll
        for (int t = 0; t < 4; ++t) bC[t] = bN[t];
    }

    // ---- epilogue: last chunk ----
#pragma unroll
    for (int t = 0; t < 4; ++t)
        acc[t] = __builtin_amdgcn_wmma_f32_16x16x32_bf16(
            false, aC.v, false, bC[t].v, (short)0, acc[t], false, false);

#pragma unroll
    for (int t = 0; t < 4; ++t) {
        const int col = tileN + t * 16 + nB;
        const float bv = BIAS ? bias[col] : 0.0f;
#pragma unroll
        for (int r = 0; r < 8; ++r) {
            const int row = tileM + mBase + r;
            float v = acc[t][r] + bv;
            if (ACT == 1) v = (v >= 0.0f) ? v : 0.01f * v;
            if (ACT == 2) v = (v >= 0.0f) ? v : 0.0f;
            C[(size_t)row * ldc + col] = v;
            if (SBF)  Cb[(size_t)row * ldcb + col] = f2bf(v);
            if (SBFT) Cb[(size_t)col * ldcb + row] = f2bf(v);
        }
    }
}

// ---------------------------------------------------------------------------
// Small helper kernels
// ---------------------------------------------------------------------------
__global__ void k_zero_u32(unsigned int* p, int n) {
    int i = blockIdx.x * 256 + threadIdx.x;
    if (i < n) p[i] = 0u;
}

__global__ void k_copy_f32(float* d, const float* s, int n) {
    int i = blockIdx.x * 256 + threadIdx.x;
    if (i < n) d[i] = s[i];
}

__global__ void k_f2bf(const float* s, __bf16* d, int n) {
    int i = blockIdx.x * 256 + threadIdx.x;
    if (i < n) d[i] = f2bf(s[i]);
}

// transpose fp32 (K x N) -> bf16 (N x K)
__global__ void k_transpose_bf(const float* __restrict__ W, __bf16* __restrict__ Wt,
                               int K, int N) {
    int idx = blockIdx.x * 256 + threadIdx.x;
    if (idx >= K * N) return;
    int k = idx / N, n = idx - k * N;
    Wt[(size_t)n * K + k] = f2bf(W[idx]);
}

// phi_x layer 1: K=2 -> scalar fp32, lrelu, bf16 out
__global__ void k_phix1(const float* __restrict__ xt, const float* __restrict__ W,
                        const float* __restrict__ bias, __bf16* __restrict__ out) {
    int idx = blockIdx.x * 256 + threadIdx.x;      // B_*HD_
    int b = idx >> 8, j = idx & 255;
    float v = xt[b * 2 + 0] * W[j] + xt[b * 2 + 1] * W[HD_ + j] + bias[j];
    v = (v >= 0.0f) ? v : 0.01f * v;
    out[idx] = f2bf(v);
}

// z = eps * exp(0.5*enc_logvar) + enc_mean  (bf16 out for phi_z GEMM)
__global__ void k_z(const float* __restrict__ eps, const float* __restrict__ em,
                    const float* __restrict__ el, __bf16* __restrict__ z) {
    int i = blockIdx.x * 256 + threadIdx.x;        // B_*ZD_
    z[i] = f2bf(eps[i] * expf(0.5f * el[i]) + em[i]);
}

// dec_mean / dec_logvar: N=2 -> scalar fp32
__global__ void k_decml(const float* __restrict__ dec,
                        const float* __restrict__ Wdm, const float* __restrict__ bdm,
                        const float* __restrict__ Wdl, const float* __restrict__ bdl,
                        float* __restrict__ dm, float* __restrict__ dl) {
    int b = blockIdx.x * 256 + threadIdx.x;
    if (b >= B_) return;
    float m0 = bdm[0], m1 = bdm[1], l0 = bdl[0], l1 = bdl[1];
    const float* row = dec + (size_t)b * HD_;
    for (int k = 0; k < HD_; ++k) {
        float a = row[k];
        m0 += a * Wdm[k * 2 + 0]; m1 += a * Wdm[k * 2 + 1];
        l0 += a * Wdl[k * 2 + 0]; l1 += a * Wdl[k * 2 + 1];
    }
    dm[b * 2 + 0] = m0; dm[b * 2 + 1] = m1;
    dl[b * 2 + 0] = l0; dl[b * 2 + 1] = l1;
}

// GRU cell (torch gate order r,z,n); writes bf16 h_new for the GCN GEMMs
__global__ void k_gru(const float* __restrict__ gi, const float* __restrict__ gh,
                      const float* __restrict__ h,
                      float* __restrict__ hn_f, __bf16* __restrict__ hn_b) {
    int idx = blockIdx.x * 256 + threadIdx.x;      // B_*RD_
    int b = idx >> 8, j = idx & 255;
    const float* gib = gi + (size_t)b * 768;
    const float* ghb = gh + (size_t)b * 768;
    float r  = 1.0f / (1.0f + expf(-(gib[j]       + ghb[j])));
    float zg = 1.0f / (1.0f + expf(-(gib[256 + j] + ghb[256 + j])));
    float n  = tanhf(gib[512 + j] + r * ghb[512 + j]);
    float out = (1.0f - zg) * n + zg * h[idx];
    hn_f[idx] = out;
    hn_b[idx] = f2bf(out);
}

// per-step KLD/NLL: one block, 1024 threads, deterministic LDS tree reduction
__global__ __launch_bounds__(1024) void k_kldnll(
    const float* __restrict__ em, const float* __restrict__ el,
    const float* __restrict__ pm, const float* __restrict__ pl,
    const float* __restrict__ xt, const float* __restrict__ dm,
    const float* __restrict__ dl, float* __restrict__ out) {
    __shared__ float sk[1024];
    __shared__ float sn[1024];
    const int b = threadIdx.x;
    float s1 = 0.0f, s2 = 0.0f, s3 = 0.0f;
    for (int j = 0; j < ZD_; ++j) {
        float e = el[b * ZD_ + j], p = pl[b * ZD_ + j];
        s1 += p - e;
        s2 += expf(e - p);
        float d = em[b * ZD_ + j] - pm[b * ZD_ + j];
        s3 += d * d * expf(-p);
    }
    float kld = 0.5f * (s1 - (float)ZD_ + s2 + s3);
    float n1 = 0.0f, n2 = 0.0f;
    for (int j = 0; j < 2; ++j) {
        float dv = dl[b * 2 + j];
        float d  = xt[b * 2 + j] - dm[b * 2 + j];
        n1 += d * d * expf(-dv);
        n2 += dv;
    }
    const float LOG2PI = 1.8378770664093453f;
    float nll = 0.5f * (n1 + 2.0f * LOG2PI + n2);
    sk[b] = kld; sn[b] = nll;
    __syncthreads();
    for (int off = 512; off > 0; off >>= 1) {
        if (b < off) { sk[b] += sk[b + off]; sn[b] += sn[b + off]; }
        __syncthreads();
    }
    if (b == 0) {
        out[0] += sk[0] * (1.0f / (float)B_);
        out[1] += sn[0] * (1.0f / (float)B_);
    }
}

// ---------------------------------------------------------------------------
// Host-side GEMM dispatch (8 fused variants)
// ---------------------------------------------------------------------------
static void gemm_launch(hipStream_t st, int code,
                        const __bf16* A, int lda, const __bf16* Bt, int ldbt, int koff,
                        const float* bias, float* C, int ldc, __bf16* Cb, int ldcb,
                        int M, int N, int K) {
    dim3 grid(M / 128, N / 64), blk(256);
    switch (code) {
    case 0: gemm_bf16_kernel<0,0,1,0,0><<<grid,blk,0,st>>>(A,lda,Bt,ldbt,koff,bias,C,ldc,Cb,ldcb,K); break; // bias, no act
    case 1: gemm_bf16_kernel<1,1,0,1,0><<<grid,blk,0,st>>>(A,lda,Bt,ldbt,koff,bias,C,ldc,Cb,ldcb,K); break; // acc + lrelu + bf
    case 2: gemm_bf16_kernel<1,0,1,1,0><<<grid,blk,0,st>>>(A,lda,Bt,ldbt,koff,bias,C,ldc,Cb,ldcb,K); break; // bias + lrelu + bf
    case 3: gemm_bf16_kernel<0,1,0,0,0><<<grid,blk,0,st>>>(A,lda,Bt,ldbt,koff,bias,C,ldc,Cb,ldcb,K); break; // acc only
    case 4: gemm_bf16_kernel<0,0,0,0,1><<<grid,blk,0,st>>>(A,lda,Bt,ldbt,koff,bias,C,ldc,Cb,ldcb,K); break; // bfT out
    case 5: gemm_bf16_kernel<2,0,1,1,0><<<grid,blk,0,st>>>(A,lda,Bt,ldbt,koff,bias,C,ldc,Cb,ldcb,K); break; // bias + relu + bf
    case 6: gemm_bf16_kernel<0,0,1,1,0><<<grid,blk,0,st>>>(A,lda,Bt,ldbt,koff,bias,C,ldc,Cb,ldcb,K); break; // bias + bf
    case 7: gemm_bf16_kernel<0,1,0,1,0><<<grid,blk,0,st>>>(A,lda,Bt,ldbt,koff,bias,C,ldc,Cb,ldcb,K); break; // acc + bf
    }
}

// ---------------------------------------------------------------------------
extern "C" void kernel_launch(void* const* d_in, const int* in_sizes, int n_in,
                              void* d_out, int out_size, void* d_ws, size_t ws_size,
                              hipStream_t stream) {
    (void)in_sizes; (void)n_in; (void)out_size; (void)ws_size;
    const float* x   = (const float*)d_in[0];   // (40,1024,2)
    const float* adj = (const float*)d_in[1];   // (40,1024,1024)
    const float* eps = (const float*)d_in[2];   // (39,1024,64)
    const float* Wpx1=(const float*)d_in[3];  const float* bpx1=(const float*)d_in[4];
    const float* Wpx2=(const float*)d_in[5];  const float* bpx2=(const float*)d_in[6];
    const float* Wpz1=(const float*)d_in[7];  const float* bpz1=(const float*)d_in[8];
    const float* Wpz2=(const float*)d_in[9];  const float* bpz2=(const float*)d_in[10];
    const float* We1 =(const float*)d_in[11]; const float* be1 =(const float*)d_in[12];
    const float* We2 =(const float*)d_in[13]; const float* be2 =(const float*)d_in[14];
    const float* Wem =(const float*)d_in[15]; const float* bem =(const float*)d_in[16];
    const float* Wel =(const float*)d_in[17]; const float* bel =(const float*)d_in[18];
    const float* Wp1 =(const float*)d_in[19]; const float* bp1 =(const float*)d_in[20];
    const float* Wp2 =(const float*)d_in[21]; const float* bp2 =(const float*)d_in[22];
    const float* Wpm =(const float*)d_in[23]; const float* bpm =(const float*)d_in[24];
    const float* Wpl =(const float*)d_in[25]; const float* bpl =(const float*)d_in[26];
    const float* Wd1 =(const float*)d_in[27]; const float* bd1 =(const float*)d_in[28];
    const float* Wd2 =(const float*)d_in[29]; const float* bd2 =(const float*)d_in[30];
    const float* Wdm =(const float*)d_in[31]; const float* bdm =(const float*)d_in[32];
    const float* Wdl =(const float*)d_in[33]; const float* bdl =(const float*)d_in[34];
    const float* Wih =(const float*)d_in[35]; const float* bih =(const float*)d_in[36];
    const float* Whh =(const float*)d_in[37]; const float* bhh =(const float*)d_in[38];
    const float* Wg1 =(const float*)d_in[39]; const float* bg1 =(const float*)d_in[40];
    const float* Wg2 =(const float*)d_in[41]; const float* bg2 =(const float*)d_in[42];
    const float* Wlg =(const float*)d_in[43]; const float* blg =(const float*)d_in[44];
    float* out = (float*)d_out;                 // [KLD, NLL, h(1024*256)]

    // deterministic bump allocator over d_ws (~25 MB total)
    char* w = (char*)d_ws;
    auto af = [&](size_t n) -> float*  { float*  p = (float*)w;  w += ((n*4 + 255) & ~(size_t)255); return p; };
    auto ab = [&](size_t n) -> __bf16* { __bf16* p = (__bf16*)w; w += ((n*2 + 255) & ~(size_t)255); return p; };

    // transposed bf16 weights (N-major)
    __bf16 *wpx2t = ab(256*256), *wpz1t = ab(256*64),  *wpz2t = ab(256*256);
    __bf16 *we1t  = ab(256*512), *we2t  = ab(256*256);
    __bf16 *wemt  = ab(64*256),  *welt  = ab(64*256);
    __bf16 *wp1t  = ab(256*256), *wp2t  = ab(256*256);
    __bf16 *wpmt  = ab(64*256),  *wplt  = ab(64*256);
    __bf16 *wd1t  = ab(256*512), *wd2t  = ab(256*256);
    __bf16 *wiht  = ab(768*512), *whht  = ab(768*256);
    __bf16 *wg1t  = ab(256*256), *wg2t  = ab(256*256), *wlgt = ab(256*512);
    // bf16 activations
    __bf16 *phix1_b = ab((size_t)B_*HD_), *phix_b = ab((size_t)B_*HD_);
    __bf16 *enc1_b  = ab((size_t)B_*HD_), *enc_b  = ab((size_t)B_*HD_);
    __bf16 *pr1_b   = ab((size_t)B_*HD_), *pr_b   = ab((size_t)B_*HD_);
    __bf16 *z_b     = ab((size_t)B_*ZD_);
    __bf16 *phiz1_b = ab((size_t)B_*HD_), *phiz_b = ab((size_t)B_*HD_);
    __bf16 *dec1_b  = ab((size_t)B_*HD_), *dec_b  = ab((size_t)B_*HD_);
    __bf16 *hn_b    = ab((size_t)B_*RD_);
    __bf16 *hWt_b   = ab((size_t)B_*256), *h1_b = ab((size_t)B_*256), *h1Wt_b = ab((size_t)B_*256);
    __bf16 *hg_b    = ab((size_t)B_*RD_), *h_b = ab((size_t)B_*RD_);
    __bf16 *adj_b   = ab((size_t)B_*B_);
    // fp32 buffers
    float *scr_C  = af((size_t)B_*HD_);                       // reusable GEMM fp32 C
    float *dec2_C = af((size_t)B_*HD_);
    float *gi_C   = af((size_t)B_*768), *gh_C = af((size_t)B_*768);
    float *encm_C = af((size_t)B_*ZD_), *encl_C = af((size_t)B_*ZD_);
    float *pm_C   = af((size_t)B_*ZD_), *pl_C   = af((size_t)B_*ZD_);
    float *dm_C   = af((size_t)B_*2),   *dl_C   = af((size_t)B_*2);
    float *h_f32  = af((size_t)B_*RD_), *hn_f32 = af((size_t)B_*RD_);

    // ---- init (stateless per call) ----
    k_zero_u32<<<(B_*RD_+255)/256,256,0,stream>>>((unsigned int*)h_f32, B_*RD_);
    k_zero_u32<<<(B_*RD_/2+255)/256,256,0,stream>>>((unsigned int*)h_b, B_*RD_/2);
    k_zero_u32<<<1,256,0,stream>>>((unsigned int*)out, 2);

    // ---- weight transpose + bf16 convert (once per launch) ----
    auto T = [&](const float* W, __bf16* Wt, int K, int N) {
        int n = K * N;
        k_transpose_bf<<<(n+255)/256,256,0,stream>>>(W, Wt, K, N);
    };
    T(Wpx2, wpx2t, 256,256);  T(Wpz1, wpz1t, 64,256);  T(Wpz2, wpz2t, 256,256);
    T(We1,  we1t,  512,256);  T(We2,  we2t,  256,256);
    T(Wem,  wemt,  256,64);   T(Wel,  welt,  256,64);
    T(Wp1,  wp1t,  256,256);  T(Wp2,  wp2t,  256,256);
    T(Wpm,  wpmt,  256,64);   T(Wpl,  wplt,  256,64);
    T(Wd1,  wd1t,  512,256);  T(Wd2,  wd2t,  256,256);
    T(Wih,  wiht,  512,768);  T(Whh,  whht,  256,768);
    T(Wg1,  wg1t,  256,256);  T(Wg2,  wg2t,  256,256);  T(Wlg, wlgt, 512,256);

    // ---- time loop: t = 1..39 (x[1:], adj[1:], eps[0:39]) ----
    for (int t = 1; t < 40; ++t) {
        const float* xt   = x   + (size_t)t * B_ * 2;
        const float* adjt = adj + (size_t)t * B_ * B_;
        const float* epst = eps + (size_t)(t - 1) * B_ * ZD_;

        k_f2bf<<<(B_*B_+255)/256,256,0,stream>>>(adjt, adj_b, B_*B_);

        // phi_x
        k_phix1<<<(B_*HD_+255)/256,256,0,stream>>>(xt, Wpx1, bpx1, phix1_b);
        gemm_launch(stream,2, phix1_b,256, wpx2t,256,0, bpx2, scr_C,256, phix_b,256, B_,256,256);
        // encoder: concat([phi_x, h]) @ We1, then We2, then mean/logvar heads
        gemm_launch(stream,0, phix_b,256, we1t,512,0,   be1,  scr_C,256, nullptr,0, B_,256,256);
        gemm_launch(stream,1, h_b,   256, we1t,512,256, nullptr, scr_C,256, enc1_b,256, B_,256,256);
        gemm_launch(stream,2, enc1_b,256, we2t,256,0,   be2,  scr_C,256, enc_b,256, B_,256,256);
        gemm_launch(stream,0, enc_b, 256, wemt,256,0,   bem,  encm_C,64, nullptr,0, B_,64,256);
        gemm_launch(stream,0, enc_b, 256, welt,256,0,   bel,  encl_C,64, nullptr,0, B_,64,256);
        // prior
        gemm_launch(stream,2, h_b,  256, wp1t,256,0, bp1, scr_C,256, pr1_b,256, B_,256,256);
        gemm_launch(stream,2, pr1_b,256, wp2t,256,0, bp2, scr_C,256, pr_b,256,  B_,256,256);
        gemm_launch(stream,0, pr_b, 256, wpmt,256,0, bpm, pm_C,64, nullptr,0, B_,64,256);
        gemm_launch(stream,0, pr_b, 256, wplt,256,0, bpl, pl_C,64, nullptr,0, B_,64,256);
        // reparameterize z, phi_z
        k_z<<<(B_*ZD_+255)/256,256,0,stream>>>(epst, encm_C, encl_C, z_b);
        gemm_launch(stream,2, z_b,    64, wpz1t,64,0,  bpz1, scr_C,256, phiz1_b,256, B_,256,64);
        gemm_launch(stream,2, phiz1_b,256,wpz2t,256,0, bpz2, scr_C,256, phiz_b,256,  B_,256,256);
        // decoder: concat([phi_z, h]) @ Wd1, Wd2, then mean/logvar (N=2 scalar)
        gemm_launch(stream,0, phiz_b,256, wd1t,512,0,   bd1, scr_C,256, nullptr,0, B_,256,256);
        gemm_launch(stream,1, h_b,   256, wd1t,512,256, nullptr, scr_C,256, dec1_b,256, B_,256,256);
        gemm_launch(stream,2, dec1_b,256, wd2t,256,0,   bd2, dec2_C,256, dec_b,256, B_,256,256);
        k_decml<<<(B_+255)/256,256,0,stream>>>(dec2_C, Wdm, bdm, Wdl, bdl, dm_C, dl_C);
        // GRU: gi = [phi_x, phi_z] @ Wih + bih ; gh = h @ Whh + bhh
        gemm_launch(stream,0, phix_b,256, wiht,512,0,   bih, gi_C,768, nullptr,0, B_,768,256);
        gemm_launch(stream,3, phiz_b,256, wiht,512,256, nullptr, gi_C,768, nullptr,0, B_,768,256);
        gemm_launch(stream,0, h_b,   256, whht,256,0,   bhh, gh_C,768, nullptr,0, B_,768,256);
        k_gru<<<(B_*RD_+255)/256,256,0,stream>>>(gi_C, gh_C, h_f32, hn_f32, hn_b);
        // GCN: h1 = relu(A (hn Wg1) + bg1); hg = A (h1 Wg2) + bg2
        gemm_launch(stream,4, hn_b,256, wg1t,256,0, nullptr, scr_C,256, hWt_b,1024, B_,256,256);
        gemm_launch(stream,5, adj_b,1024, hWt_b,1024,0, bg1, scr_C,256, h1_b,256, B_,256,1024);
        gemm_launch(stream,4, h1_b,256, wg2t,256,0, nullptr, scr_C,256, h1Wt_b,1024, B_,256,256);
        gemm_launch(stream,6, adj_b,1024, h1Wt_b,1024,0, bg2, scr_C,256, hg_b,256, B_,256,1024);
        // h_new = [h_g, h_gru] @ Wlg + blg   (overwrites carry h)
        gemm_launch(stream,0, hg_b,256, wlgt,512,0,   blg, h_f32,256, nullptr,0, B_,256,256);
        gemm_launch(stream,7, hn_b,256, wlgt,512,256, nullptr, h_f32,256, h_b,256, B_,256,256);
        // losses
        k_kldnll<<<1,1024,0,stream>>>(encm_C, encl_C, pm_C, pl_C, xt, dm_C, dl_C, out);
    }

    // final hidden state -> d_out[2:]
    k_copy_f32<<<(B_*RD_+255)/256,256,0,stream>>>(out + 2, h_f32, B_*RD_);
}
